// GaussianGradientGRUPolicy_27324581937665
// MI455X (gfx1250) — compile-verified, matrix operand order checked by reading, not run on hardware
//
#include <hip/hip_runtime.h>
#include <hip/hip_bf16.h>
#include <stdint.h>

// Problem dims (fixed by the reference): T=512, B=256, F=64, H=256, A=8
#define TT 512
#define BB 256
#define FF 64
#define HH 256
#define H3 (3 * HH)
#define HP 264   // padded LDS row stride (halves) to break ds bank conflicts

typedef __attribute__((ext_vector_type(16))) __bf16 v16bf;
typedef __attribute__((ext_vector_type(8)))  float  v8f;

union BFrag { uint32_t u[8]; uint4 q[2]; v16bf v; };

// Hardware f32->bf16 (RNE) conversion; clang lowers the cast to v_cvt*_bf16_f32.
__device__ __forceinline__ uint16_t f2bf(float f) {
  union { __bf16 b; uint16_t u; } c; c.b = (__bf16)f;
  return c.u;
}
__device__ __forceinline__ uint32_t f2bf2(float lo, float hi) {
  return (uint32_t)f2bf(lo) | ((uint32_t)f2bf(hi) << 16);
}
__device__ __forceinline__ float bf2f(uint16_t h) {
  union { uint32_t u; float f; } c; c.u = (uint32_t)h << 16;
  return c.f;
}
// K index of the low bf16 held in VGPR v for lane-group grp (16-bit operand
// layout per CDNA5 ISA 7.12.2: grp0 holds K 0..7 / 16..23, grp1 K 8..15 / 24..31)
__device__ __forceinline__ int kpair_lo(int v, int grp) {
  return 2 * v + 8 * grp + ((v >= 4) ? 8 : 0);
}
__device__ __forceinline__ v8f bf16_wmma(v16bf a, v16bf b, v8f c) {
  return __builtin_amdgcn_wmma_f32_16x16x32_bf16(false, a, false, b, (short)0, c,
                                                 false, false);
}
// B fragment (K=32 slab of one 16-wide N tile) from the packed weight image:
// packed[((nt*nK + ks)*32 + lane)*8 + v] -> two global_load_b128 per lane.
__device__ __forceinline__ v16bf load_bfrag(const uint32_t* __restrict__ p,
                                            int nt, int ks, int nK, int lane) {
  BFrag f;
  const uint32_t* q = p + (((size_t)nt * nK + ks) * 32 + lane) * 8;
  f.q[0] = *(const uint4*)q;
  f.q[1] = *(const uint4*)(q + 4);
  return f.v;
}
// A fragment (16 rows x K=32) from a row-major bf16 LDS tile: two ds_load_b128.
__device__ __forceinline__ v16bf lds_afrag(const uint16_t* base, int stride,
                                           int mrow, int kbase, int ln, int grp) {
  BFrag f;
  const uint16_t* r = base + (size_t)(mrow + ln) * stride + kbase + 8 * grp;
  f.q[0] = *(const uint4*)r;         // K = kbase+8g .. +7      (v0..v3)
  f.q[1] = *(const uint4*)(r + 16);  // K = kbase+16+8g .. +7   (v4..v7)
  return f.v;
}
__device__ __forceinline__ float sigmoid_f(float x) {
  return 1.0f / (1.0f + __expf(-x));
}

// ---------------------------------------------------------------------------
// Pack a row-major fp32 weight W[N][K] (used as B = W^T, i.e. out = in @ W^T)
// into per-lane WMMA-B bf16 fragment order.
// ---------------------------------------------------------------------------
__global__ void pack_w_bf16(const float* __restrict__ W, uint32_t* __restrict__ outp,
                            int N, int K) {
  int nK = K >> 5;
  int total = (N >> 4) * nK * 32 * 8;
  int i = blockIdx.x * blockDim.x + threadIdx.x;
  if (i >= total) return;
  int v    = i & 7;
  int lane = (i >> 3) & 31;
  int rest = i >> 8;            // nt*nK + ks
  int ks   = rest % nK;
  int nt   = rest / nK;
  int n    = nt * 16 + (lane & 15);
  int grp  = lane >> 4;
  int k    = ks * 32 + kpair_lo(v, grp);
  outp[i] = f2bf2(W[(size_t)n * K + k], W[(size_t)n * K + k + 1]);
}

// ---------------------------------------------------------------------------
// Fused: sf = leaky_relu(x @ W1^T + b1) (kept in LDS, never hits HBM), then
// gi = sf @ W_ih^T + b_ih, stored to HBM in WMMA C-fragment order (bf16) so the
// scan kernel reads each gate tile back with one b128 load per lane.
// One block = 64 consecutive (t,b) rows x all 768 gi columns. 256 thr / 8 waves.
// ---------------------------------------------------------------------------
__global__ __launch_bounds__(256) void sf_gi_kernel(
    const float* __restrict__ x, const float* __restrict__ b1,
    const float* __restrict__ b_ih, const uint32_t* __restrict__ pW1,
    const uint32_t* __restrict__ pWih, uint16_t* __restrict__ giFrag) {
  __shared__ __align__(16) uint16_t sfLds[64 * HP];
  const int wave = threadIdx.x >> 5;
  const int lane = threadIdx.x & 31;
  const int ln = lane & 15, grp = lane >> 4;
  const size_t rowBase = (size_t)blockIdx.x * 64;

  // ---- stage 1: sf tile (M=64, N=256, K=64), A converted f32->bf16 on load --
  for (int mt = 0; mt < 4; ++mt) {
    BFrag a[2];
    const size_t row = rowBase + mt * 16 + ln;
#pragma unroll
    for (int ks = 0; ks < 2; ++ks) {
#pragma unroll
      for (int v = 0; v < 8; ++v) {
        int k = ks * 32 + kpair_lo(v, grp);
        float2 xv = *(const float2*)(x + row * FF + k);
        a[ks].u[v] = f2bf2(xv.x, xv.y);
      }
    }
#pragma unroll
    for (int j = 0; j < 2; ++j) {
      int nt = wave * 2 + j;                       // 16 N-tiles over 8 waves
      v8f acc = {};
      acc = bf16_wmma(a[0].v, load_bfrag(pW1, nt, 0, 2, lane), acc);
      acc = bf16_wmma(a[1].v, load_bfrag(pW1, nt, 1, 2, lane), acc);
      int col = nt * 16 + ln;
      float bias = b1[col];
#pragma unroll
      for (int v = 0; v < 8; ++v) {
        int r = mt * 16 + v + 8 * grp;
        float s = acc[v] + bias;
        s = (s >= 0.0f) ? s : (0.01f * s);         // LeakyReLU
        sfLds[r * HP + col] = f2bf(s);
      }
    }
  }
  __syncthreads();

  // ---- stage 2: gi tile (M=64, N=768, K=256), A from LDS -------------------
  const size_t gtBase = rowBase >> 4;              // global 16-row tile index
  for (int mt = 0; mt < 4; ++mt) {
    v16bf a[8];
#pragma unroll
    for (int ks = 0; ks < 8; ++ks)
      a[ks] = lds_afrag(sfLds, HP, mt * 16, ks * 32, ln, grp);
    for (int j = 0; j < 6; ++j) {
      int nt = wave * 6 + j;                       // 48 N-tiles over 8 waves
      v8f acc = {};
#pragma unroll
      for (int ks = 0; ks < 8; ++ks)
        acc = bf16_wmma(a[ks], load_bfrag(pWih, nt, ks, 8, lane), acc);
      float bias = b_ih[nt * 16 + ln];
      union { uint32_t u[4]; uint4 q; } o;
#pragma unroll
      for (int v = 0; v < 4; ++v)
        o.u[v] = f2bf2(acc[2 * v] + bias, acc[2 * v + 1] + bias);
      *(uint4*)(giFrag + (((gtBase + mt) * 48 + nt) * 32 + lane) * 8) = o.q;
    }
  }
}

// ---------------------------------------------------------------------------
// Persistent single-workgroup GRU scan. 512 threads = 16 waves (wave32).
// h double-buffered in LDS (bf16, padded rows): hA = done-masked h (A operand
// + gate input), hB = raw h_new. Wave w owns H columns [16w,16w+16) and hence
// the matching r/z/n gh tiles -> gate math fuses on the three accumulators.
// Heads (tanh / softplus) computed on h_final at the end.
// ---------------------------------------------------------------------------
__global__ __launch_bounds__(512) void gru_scan_kernel(
    const int* __restrict__ dones, const uint16_t* __restrict__ giFrag,
    const uint32_t* __restrict__ pWhh, const float* __restrict__ b_hh,
    const float* __restrict__ Wm, const float* __restrict__ bm,
    const float* __restrict__ Ws, const float* __restrict__ bs,
    float* __restrict__ out) {
  __shared__ __align__(16) uint16_t hbuf[2 * BB * HP];   // 264 KB LDS
  uint16_t* hA = hbuf;
  uint16_t* hB = hbuf + BB * HP;
  const int tid = threadIdx.x;
  const int wave = tid >> 5, lane = tid & 31;
  const int ln = lane & 15, grp = lane >> 4;
  const int c0 = wave * 16;

  for (int i = tid; i < BB * (HP / 2); i += 512) ((uint32_t*)hB)[i] = 0u;  // h0
  __syncthreads();

  const float bhr = b_hh[c0 + ln];
  const float bhz = b_hh[HH + c0 + ln];
  const float bhn = b_hh[2 * HH + c0 + ln];

  for (int t = 0; t < TT; ++t) {
    // done-mask pass: hA = dones[t,b] ? 0 : hB
    const int* dr = dones + (size_t)t * BB;
    for (int i = tid; i < BB * (HP / 2); i += 512) {
      int b = i / (HP / 2);
      ((uint32_t*)hA)[i] = (dr[b] > 0) ? 0u : ((const uint32_t*)hB)[i];
    }
    __syncthreads();

    const size_t gT = (size_t)t * 16;
    for (int mt = 0; mt < 16; ++mt) {
      const size_t gb = (gT + mt) * 48;
      if (t + 1 < TT) {  // pull next step's gi fragments toward L2/L0
        __builtin_prefetch(giFrag + ((gb + 16 * 48 + wave) * 32 + lane) * 8, 0, 1);
        __builtin_prefetch(giFrag + ((gb + 16 * 48 + 16 + wave) * 32 + lane) * 8, 0, 1);
        __builtin_prefetch(giFrag + ((gb + 16 * 48 + 32 + wave) * 32 + lane) * 8, 0, 1);
      }
      v16bf a[8];
#pragma unroll
      for (int ks = 0; ks < 8; ++ks)
        a[ks] = lds_afrag(hA, HP, mt * 16, ks * 32, ln, grp);
      v8f ar = {}, az = {}, an = {};
#pragma unroll
      for (int ks = 0; ks < 8; ++ks) {
        ar = bf16_wmma(a[ks], load_bfrag(pWhh, wave,      ks, 8, lane), ar);
        az = bf16_wmma(a[ks], load_bfrag(pWhh, 16 + wave, ks, 8, lane), az);
        an = bf16_wmma(a[ks], load_bfrag(pWhh, 32 + wave, ks, 8, lane), an);
      }
      union { uint16_t h[8]; uint4 q; } gr, gz, gn;
      gr.q = *(const uint4*)(giFrag + ((gb + wave) * 32 + lane) * 8);
      gz.q = *(const uint4*)(giFrag + ((gb + 16 + wave) * 32 + lane) * 8);
      gn.q = *(const uint4*)(giFrag + ((gb + 32 + wave) * 32 + lane) * 8);
#pragma unroll
      for (int v = 0; v < 8; ++v) {
        int row = mt * 16 + v + 8 * grp;           // batch index b
        float hu = bf2f(hA[row * HP + c0 + ln]);   // masked h (reference semantics)
        float r = sigmoid_f(bf2f(gr.h[v]) + ar[v] + bhr);
        float z = sigmoid_f(bf2f(gz.h[v]) + az[v] + bhz);
        float n = tanhf(bf2f(gn.h[v]) + r * (an[v] + bhn));
        float hn = (1.0f - z) * n + z * hu;
        hB[row * HP + c0 + ln] = f2bf(hn);
        if (t == TT - 1)                           // hidden output, fp32
          out[4096 + (size_t)row * HH + c0 + ln] = hn;
      }
    }
    __syncthreads();
  }

  // heads on h_final (== outs[-1]): means = tanh(h Wm^T + bm),
  // stds = max(softplus(h Ws^T + bs), 1e-5). 4096 outputs over 512 threads.
  for (int i = 0; i < 8; ++i) {
    int oid = tid + 512 * i;
    int head = oid >> 11;          // 0 = mean, 1 = std
    int rem = oid & 2047;
    int b = rem >> 3, ai = rem & 7;
    const float* W = head ? Ws : Wm;
    float acc = head ? bs[ai] : bm[ai];
    const uint16_t* hr = hB + (size_t)b * HP;
    for (int jj = 0; jj < HH; ++jj) acc += bf2f(hr[jj]) * W[ai * HH + jj];
    float o;
    if (head == 0) {
      o = tanhf(acc);
    } else {
      float sp = (acc > 20.0f) ? acc : log1pf(__expf(acc));
      o = fmaxf(sp, 1e-5f);
    }
    out[oid] = o;
  }
}

// ---------------------------------------------------------------------------
// Host launcher. Workspace layout (bytes):
//   [0,       32K)   packed W1   (N=256,K=64)
//   [32K,    416K)   packed W_ih (N=768,K=256)
//   [416K,   800K)   packed W_hh (N=768,K=256)
//   [800K, 800K+192M) gi fragments, bf16, C-fragment order
// ---------------------------------------------------------------------------
extern "C" void kernel_launch(void* const* d_in, const int* in_sizes, int n_in,
                              void* d_out, int out_size, void* d_ws, size_t ws_size,
                              hipStream_t stream) {
  const float* x     = (const float*)d_in[0];
  const int*   dones = (const int*)d_in[1];
  const float* W1    = (const float*)d_in[2];
  const float* b1    = (const float*)d_in[3];
  const float* W_ih  = (const float*)d_in[4];
  const float* W_hh  = (const float*)d_in[5];
  const float* b_ih  = (const float*)d_in[6];
  const float* b_hh  = (const float*)d_in[7];
  const float* Wm    = (const float*)d_in[8];
  const float* bm    = (const float*)d_in[9];
  const float* Ws    = (const float*)d_in[10];
  const float* bs    = (const float*)d_in[11];
  float* out = (float*)d_out;

  uint8_t* ws = (uint8_t*)d_ws;
  uint32_t* pW1    = (uint32_t*)(ws);
  uint32_t* pWih   = (uint32_t*)(ws + (32u << 10));
  uint32_t* pWhh   = (uint32_t*)(ws + (32u << 10) + (384u << 10));
  uint16_t* giFrag = (uint16_t*)(ws + (32u << 10) + 2u * (384u << 10));

  pack_w_bf16<<<(8192 + 255) / 256, 256, 0, stream>>>(W1, pW1, HH, FF);
  pack_w_bf16<<<(98304 + 255) / 256, 256, 0, stream>>>(W_ih, pWih, H3, HH);
  pack_w_bf16<<<(98304 + 255) / 256, 256, 0, stream>>>(W_hh, pWhh, H3, HH);

  sf_gi_kernel<<<(TT * BB) / 64, 256, 0, stream>>>(x, b1, b_ih, pW1, pWih, giFrag);

  gru_scan_kernel<<<1, 512, 0, stream>>>(dones, giFrag, pWhh, b_hh,
                                         Wm, bm, Ws, bs, out);
}
